// AttentionReadout_10771777978557
// MI455X (gfx1250) — matrix operand order repaired and checked by reading, not compiled, not run on hardware
//
#include <hip/hip_runtime.h>
#include <hip/hip_bf16.h>

// MI455X / gfx1250: wave32, WMMA 16x16x32 f16 -> f32, async global->LDS copies.
// One workgroup (8 waves) per graph of 64 nodes; everything stays in LDS.

typedef _Float16 half8  __attribute__((ext_vector_type(8)));
typedef _Float16 half16 __attribute__((ext_vector_type(16)));
typedef float    floatx8 __attribute__((ext_vector_type(8)));

// Async-copy builtin pointer types: param1 = int4 addrspace(1)*, param2 = int4 addrspace(3)*
typedef int v4i __attribute__((vector_size(16)));
typedef __attribute__((address_space(1))) v4i* gv4i_ptr;
typedef __attribute__((address_space(3))) v4i* lv4i_ptr;
typedef __attribute__((address_space(1))) void* gas_vptr;
typedef __attribute__((address_space(3))) void* lds_vptr;

#define TO_G128(p) ((gv4i_ptr)(gas_vptr)(void*)(p))
#define TO_L128(p) ((lv4i_ptr)(lds_vptr)(void*)(p))

#if defined(__has_builtin)
#if __has_builtin(__builtin_amdgcn_global_load_async_to_lds_b128) && \
    __has_builtin(__builtin_amdgcn_s_wait_asynccnt)
#define HAVE_ASYNC_LDS 1
#endif
#if __has_builtin(__builtin_amdgcn_global_store_async_from_lds_b128)
#define HAVE_ASYNC_ST 1
#endif
#endif

#define WMMA_F16(a, b, c) \
    __builtin_amdgcn_wmma_f32_16x16x32_f16(false, (a), false, (b), (short)0, (c), false, false)

// ---- fragment loaders (contiguous k order per lane half, same for A and B) -------
// A matrix 16x32 (MxK), row-major source with row pitch `pitch` (in halves):
//   lane L (0..31): M = m0 + (L&15); halves 0..15 -> K = k0 + (L>>4)*16 + 0..15
__device__ __forceinline__ half16 load_frag_a(const _Float16* src, int pitch,
                                              int m0, int k0, int lane) {
    int m   = m0 + (lane & 15);
    int off = k0 + ((lane >> 4) << 4);
    union { half16 v; half8 h[2]; } u;
    u.h[0] = *(const half8*)(src + m * pitch + off);
    u.h[1] = *(const half8*)(src + m * pitch + off + 8);
    return u.v;
}

// B matrix 32x16 (KxN), source stored as Bt[N][K] row-major, pitch in halves:
//   lane L: N = n0 + (L&15); halves 0..15 -> K = k0 + (L>>4)*16 + 0..15
__device__ __forceinline__ half16 load_frag_b(const _Float16* src, int pitch,
                                              int n0, int k0, int lane) {
    int n   = n0 + (lane & 15);
    int off = k0 + ((lane >> 4) << 4);
    return *(const half16*)(src + n * pitch + off);
}

// ---- weight conversion: fp32 -> f16 workspace copies ----------------------
__global__ void convert_w_kernel(const float* __restrict__ src,
                                 _Float16* __restrict__ dst, int n) {
    int i = blockIdx.x * blockDim.x + threadIdx.x;
    if (i < n) dst[i] = (_Float16)src[i];
}

// ---- fused per-graph kernel ----------------------------------------------
// LDS map (262144 bytes dynamic):
//   phase 0:  [0,131072) f32 x staging (async),  [196608,262144) xh f16
//   [0      , 65536 )  qh   [64][512] f16   -> reused as ctx [64][512] f16
//   [65536  , 131072)  kh   [64][512] f16   \__ reused as aout [64][512] f32
//   [131072 , 196608)  vT   [512][64] f16   /
//   [196608 , 262144)  xh   [64][512] f16   -> reused as attn [8][64][64] f16,
//                                             then gates[64] + out staging f32
__global__ __launch_bounds__(256)
void attn_readout_fused(const float* __restrict__ x,
                        const float* __restrict__ in_proj_b,
                        const float* __restrict__ out_proj_b,
                        const float* __restrict__ gate_w,
                        const float* __restrict__ gate_b,
                        const _Float16* __restrict__ wq,   // [1536][512] f16
                        const _Float16* __restrict__ wo,   // [512][512]  f16
                        float* __restrict__ out) {
    extern __shared__ char smem[];
    _Float16* qh      = (_Float16*)(smem);
    _Float16* kh      = (_Float16*)(smem + 65536);
    _Float16* vT      = (_Float16*)(smem + 131072);
    _Float16* xh      = (_Float16*)(smem + 196608);
    _Float16* attnbuf = xh;                 // 8 heads * 64*64 f16
    _Float16* ctx     = qh;                 // [64][512] f16
    float*    aout    = (float*)(smem + 65536);     // [64][512] f32
    float*    gates   = (float*)(smem + 196608);    // [64] f32
    float*    ostage  = (float*)(smem + 196608 + 256); // [512] f32

    const int g    = blockIdx.x;
    const int tid  = threadIdx.x;
    const int lane = tid & 31;
    const int wave = tid >> 5;              // 8 waves; wave == head in attn phases

    // Warm L2/WGP$ for the f16 weights this block is about to stream.
    __builtin_prefetch(wq + (size_t)tid * 3072, 0, 3);
    __builtin_prefetch(wo + (size_t)tid * 1024, 0, 3);

    // ---- Phase 0: stage x tile into LDS (async DMA), convert to f16 ------
    const float* xg = x + (size_t)g * 64 * 512;
#if HAVE_ASYNC_LDS
    {
        float* xstage = (float*)smem;       // 128 KB f32 staging in q/k region
        for (int i = tid * 4; i < 64 * 512; i += 256 * 4) {
            __builtin_amdgcn_global_load_async_to_lds_b128(
                TO_G128(xg + i), TO_L128(xstage + i), 0, 0);
        }
        __builtin_amdgcn_s_wait_asynccnt(0);
        __syncthreads();
        for (int i = tid; i < 64 * 512; i += 256) xh[i] = (_Float16)xstage[i];
    }
#else
    for (int i = tid; i < 64 * 512; i += 256) xh[i] = (_Float16)xg[i];
#endif
    __syncthreads();

    // ---- Phase 1: QKV = x @ Wqkv^T  (64x1536, K=512) ---------------------
    // 4 M-tiles x 96 N-tiles = 384 tiles over 8 waves.
    for (int t = wave; t < 384; t += 8) {
        int mt = t & 3, nt = t >> 2;
        floatx8 c = {};
        for (int ks = 0; ks < 16; ++ks) {
            half16 a = load_frag_a(xh, 512, mt * 16, ks * 32, lane);
            half16 b = load_frag_b(wq, 512, nt * 16, ks * 32, lane);
            c = WMMA_F16(a, b, c);
        }
        int e    = nt * 16 + (lane & 15);
        float bs = in_proj_b[e];
        int mrow = mt * 16 + ((lane >> 4) << 3);
        for (int r = 0; r < 8; ++r) {
            float v = c[r] + bs;
            int m = mrow + r;
            if (e < 512)       qh[m * 512 + e]            = (_Float16)v;
            else if (e < 1024) kh[m * 512 + (e - 512)]    = (_Float16)v;
            else               vT[(e - 1024) * 64 + m]    = (_Float16)v;  // V transposed
        }
    }
    __syncthreads();

    // ---- Phase 2: scores = (Q @ K^T) / sqrt(hd), one head per wave -------
    _Float16* attn = attnbuf + wave * 64 * 64;
    for (int t = 0; t < 16; ++t) {
        int mt = t & 3, nt = t >> 2;
        floatx8 c = {};
        for (int ks = 0; ks < 2; ++ks) {
            half16 a = load_frag_a(qh, 512, mt * 16, wave * 64 + ks * 32, lane);
            half16 b = load_frag_b(kh, 512, nt * 16, wave * 64 + ks * 32, lane);
            c = WMMA_F16(a, b, c);
        }
        int n    = nt * 16 + (lane & 15);
        int mrow = mt * 16 + ((lane >> 4) << 3);
        for (int r = 0; r < 8; ++r)
            attn[(mrow + r) * 64 + n] = (_Float16)(c[r] * 0.125f);
    }
    __syncthreads();

    // ---- softmax over each of the 8*64 rows (2 rows/thread) --------------
    for (int row = tid; row < 512; row += 256) {
        _Float16* p = attnbuf + row * 64;
        float mx = -3.0e38f;
        for (int j = 0; j < 64; ++j) mx = fmaxf(mx, (float)p[j]);
        float s = 0.f;
        for (int j = 0; j < 64; ++j) s += __expf((float)p[j] - mx);
        float inv = 1.f / s;
        for (int j = 0; j < 64; ++j)
            p[j] = (_Float16)(__expf((float)p[j] - mx) * inv);
    }
    __syncthreads();

    // ---- Phase 3: ctx = attn @ V (per head), into qh region as f16 -------
    for (int t = 0; t < 16; ++t) {
        int mt = t & 3, dt = t >> 2;
        floatx8 c = {};
        for (int ks = 0; ks < 2; ++ks) {
            half16 a = load_frag_a(attn, 64, mt * 16, ks * 32, lane);
            half16 b = load_frag_b(vT, 64, wave * 64 + dt * 16, ks * 32, lane);
            c = WMMA_F16(a, b, c);
        }
        int d    = wave * 64 + dt * 16 + (lane & 15);
        int mrow = mt * 16 + ((lane >> 4) << 3);
        for (int r = 0; r < 8; ++r)
            ctx[(mrow + r) * 512 + d] = (_Float16)c[r];
    }
    __syncthreads();

    // ---- Phase 4: attn_out = ctx @ Wout^T + b  (64x512, K=512) -----------
    for (int t = wave; t < 128; t += 8) {
        int mt = t & 3, nt = t >> 2;
        floatx8 c = {};
        for (int ks = 0; ks < 16; ++ks) {
            half16 a = load_frag_a(ctx, 512, mt * 16, ks * 32, lane);
            half16 b = load_frag_b(wo, 512, nt * 16, ks * 32, lane);
            c = WMMA_F16(a, b, c);
        }
        int e    = nt * 16 + (lane & 15);
        float bs = out_proj_b[e];
        int mrow = mt * 16 + ((lane >> 4) << 3);
        for (int r = 0; r < 8; ++r)
            aout[(mrow + r) * 512 + e] = c[r] + bs;
    }
    __syncthreads();

    // ---- Phase 5: gates = sigmoid(attn_out @ gate_w + b); weighted sum ---
    if (tid < 64) {
        float s = gate_b[0];
        for (int d = 0; d < 512; ++d) s += aout[tid * 512 + d] * gate_w[d];
        gates[tid] = 1.f / (1.f + __expf(-s));
    }
    __syncthreads();
#if HAVE_ASYNC_LDS && HAVE_ASYNC_ST
    for (int d = tid; d < 512; d += 256) {
        float s = 0.f;
        for (int n = 0; n < 64; ++n) s += aout[n * 512 + d] * gates[n];
        ostage[d] = s;
    }
    __syncthreads();
    if (tid < 128) {
        int d = tid * 4;
        __builtin_amdgcn_global_store_async_from_lds_b128(
            TO_G128(out + (size_t)g * 512 + d), TO_L128(ostage + d), 0, 0);
    }
    __builtin_amdgcn_s_wait_asynccnt(0);
#else
    for (int d = tid; d < 512; d += 256) {
        float s = 0.f;
        for (int n = 0; n < 64; ++n) s += aout[n * 512 + d] * gates[n];
        out[(size_t)g * 512 + d] = s;
    }
#endif
}

extern "C" void kernel_launch(void* const* d_in, const int* in_sizes, int n_in,
                              void* d_out, int out_size, void* d_ws, size_t ws_size,
                              hipStream_t stream) {
    const float* x          = (const float*)d_in[0];
    // d_in[1] = batch ids (int64) -- graphs are uniform 64 nodes, unused
    const float* in_proj_w  = (const float*)d_in[2];   // [1536][512]
    const float* in_proj_b  = (const float*)d_in[3];
    const float* out_proj_w = (const float*)d_in[4];   // [512][512]
    const float* out_proj_b = (const float*)d_in[5];
    const float* gate_w     = (const float*)d_in[6];   // [1][512]
    const float* gate_b     = (const float*)d_in[7];
    float* out = (float*)d_out;

    const int nq = in_sizes[2];            // 1536*512
    const int no = in_sizes[4];            // 512*512
    _Float16* wq = (_Float16*)d_ws;
    _Float16* wo = (_Float16*)((char*)d_ws + (size_t)nq * sizeof(_Float16));

    convert_w_kernel<<<(nq + 255) / 256, 256, 0, stream>>>(in_proj_w, wq, nq);
    convert_w_kernel<<<(no + 255) / 256, 256, 0, stream>>>(out_proj_w, wo, no);

    const int total = in_sizes[0] / 512;
    const int G     = total / 64;          // 1024 graphs
    attn_readout_fused<<<G, 256, 262144, stream>>>(
        x, in_proj_b, out_proj_b, gate_w, gate_b, wq, wo, out);
}
